// GNN_28226525070233
// MI455X (gfx1250) — matrix operand order repaired and checked by reading, not compiled, not run on hardware
//
#include <hip/hip_runtime.h>
#include <hip/hip_bf16.h>

#define D    128
#define H1   256
#define D3   384
#define ROWS 64
#define NUM_HOPS 4

typedef __attribute__((ext_vector_type(16))) __bf16 v16bf;
typedef __attribute__((ext_vector_type(8)))  float  v8f;

// ---------------------------------------------------------------------------
// WMMA helpers (CDNA5 16x16x32 bf16, f32 accumulate)
// ---------------------------------------------------------------------------
__device__ __forceinline__ v8f wmma_bf16(v16bf a, v16bf b, v8f c) {
  return __builtin_amdgcn_wmma_f32_16x16x32_bf16(false, a, false, b, (short)0, c,
                                                 false, false);
}

// A fragment: 16x32 bf16 tile from a row-major source (row stride = `stride`).
// Layout: lanes 0-15 row M=lane, K = 0..15; lanes 16-31 row
// M=lane-16, K = 16..31.
__device__ __forceinline__ v16bf load_A_lds(const __bf16* src, int stride, int lane) {
  const int row = lane & 15;
  const int kb  = (lane >> 4) << 4;
  const __bf16* r = src + row * stride;
  v16bf a;
#pragma unroll
  for (int e = 0; e < 16; ++e) a[e] = r[kb + e];
  return a;
}

// B fragment: 32x16 tile of W (KxN) stored TRANSPOSED row-major Wt[N][K].
// Lanes 0-15: col N=lane, K=0..15; lanes 16-31: col N=lane-16, K=16..31.
// Contiguous 32B per lane -> global_load_b128 pairs, L2-resident weights.
__device__ __forceinline__ v16bf load_B_t(const __bf16* src, int ldk, int lane) {
  const int col = lane & 15;
  const int kb  = (lane >> 4) << 4;
  const __bf16* r = src + (size_t)col * ldk + kb;
  v16bf b;
#pragma unroll
  for (int e = 0; e < 16; ++e) b[e] = r[e];
  return b;
}

// ---------------------------------------------------------------------------
// Utility kernels
// ---------------------------------------------------------------------------
__global__ void k_convert_w(const float* __restrict__ W, __bf16* __restrict__ Wt,
                            int K, int N) {
  int i = blockIdx.x * blockDim.x + threadIdx.x;
  if (i >= K * N) return;
  int k = i / N, n = i % N;
  Wt[(size_t)n * K + k] = (__bf16)W[i];
}

__global__ void k_zero(float* __restrict__ p, long n) {
  long i = (long)blockIdx.x * blockDim.x + threadIdx.x;
  if (i < n) p[i] = 0.f;
}

__global__ void k_count(const int* __restrict__ pidx, const int* __restrict__ cidx,
                        float* __restrict__ cp, float* __restrict__ cc, int nE) {
  int i = blockIdx.x * blockDim.x + threadIdx.x;
  if (i < nE) {
    atomicAdd(&cp[pidx[i]], 1.f);
    atomicAdd(&cc[cidx[i]], 1.f);
  }
}

__global__ void k_inv(float* __restrict__ cp, float* __restrict__ cc, int n) {
  int i = blockIdx.x * blockDim.x + threadIdx.x;
  if (i < n) {
    cp[i] = 1.f / fmaxf(cp[i], 1.f);
    cc[i] = 1.f / fmaxf(cc[i], 1.f);
  }
}

// ---------------------------------------------------------------------------
// MLP_V: h = relu(relu(X@Vw1+b1)@Vw2+b2).  64 rows / WG, 8 waves.
// ---------------------------------------------------------------------------
__global__ __launch_bounds__(256)
void k_mlp_v(const float* __restrict__ X,
             const __bf16* __restrict__ W1t, const float* __restrict__ b1,
             const __bf16* __restrict__ W2t, const float* __restrict__ b2,
             float* __restrict__ h_f32, __bf16* __restrict__ h_bf, int nRows) {
  __shared__ __bf16 sX[ROWS][D];
  __shared__ __bf16 sH[ROWS][H1];
  const int tid  = threadIdx.x;
  const int lane = tid & 31;
  const int wave = tid >> 5;
  const int col  = lane & 15;
  const int mrow = (lane >> 4) * 8;
  const int row0 = blockIdx.x * ROWS;

  for (int i = tid; i < ROWS * (D / 4); i += 256) {
    int r = i / (D / 4), c4 = i % (D / 4);
    int g = row0 + r;
    float4 v = {0.f, 0.f, 0.f, 0.f};
    if (g < nRows) v = ((const float4*)(X + (size_t)g * D))[c4];
    sX[r][c4 * 4 + 0] = (__bf16)v.x;
    sX[r][c4 * 4 + 1] = (__bf16)v.y;
    sX[r][c4 * 4 + 2] = (__bf16)v.z;
    sX[r][c4 * 4 + 3] = (__bf16)v.w;
  }
  __syncthreads();

  // layer 1: [64,256]; wave: 2 col tiles x 4 row tiles, K=128
#pragma unroll
  for (int tc = 0; tc < 2; ++tc) {
    const int n0 = (wave * 2 + tc) * 16;
    v8f acc[4] = {};
#pragma unroll
    for (int ks = 0; ks < D / 32; ++ks) {
      v16bf b = load_B_t(W1t + (size_t)n0 * D + ks * 32, D, lane);
#pragma unroll
      for (int rt = 0; rt < 4; ++rt) {
        v16bf a = load_A_lds(&sX[rt * 16][0] + ks * 32, D, lane);
        acc[rt] = wmma_bf16(a, b, acc[rt]);
      }
    }
    const float bias = b1[n0 + col];
#pragma unroll
    for (int rt = 0; rt < 4; ++rt)
#pragma unroll
      for (int r = 0; r < 8; ++r)
        sH[rt * 16 + mrow + r][n0 + col] = (__bf16)fmaxf(acc[rt][r] + bias, 0.f);
  }
  __syncthreads();

  // layer 2: [64,128]; wave: 1 col tile x 4 row tiles, K=256
  {
    const int n0 = wave * 16;
    v8f acc[4] = {};
#pragma unroll
    for (int ks = 0; ks < H1 / 32; ++ks) {
      v16bf b = load_B_t(W2t + (size_t)n0 * H1 + ks * 32, H1, lane);
#pragma unroll
      for (int rt = 0; rt < 4; ++rt) {
        v16bf a = load_A_lds(&sH[rt * 16][0] + ks * 32, H1, lane);
        acc[rt] = wmma_bf16(a, b, acc[rt]);
      }
    }
    const float bias = b2[n0 + col];
#pragma unroll
    for (int rt = 0; rt < 4; ++rt)
#pragma unroll
      for (int r = 0; r < 8; ++r) {
        int g = row0 + rt * 16 + mrow + r;
        if (g < nRows) {
          float v = fmaxf(acc[rt][r] + bias, 0.f);
          h_f32[(size_t)g * D + n0 + col] = v;
          h_bf [(size_t)g * D + n0 + col] = (__bf16)v;
        }
      }
  }
}

// ---------------------------------------------------------------------------
// MLP_E: layer 1 elementwise (scalar indicator), layer 2 WMMA GEMM.
// ---------------------------------------------------------------------------
__global__ __launch_bounds__(256)
void k_mlp_e(const float* __restrict__ ind,
             const float* __restrict__ w1, const float* __restrict__ b1,
             const __bf16* __restrict__ W2t, const float* __restrict__ b2,
             __bf16* __restrict__ out, int nRows) {
  __shared__ float  sx[ROWS];
  __shared__ __bf16 sH[ROWS][H1];
  const int tid  = threadIdx.x;
  const int lane = tid & 31;
  const int wave = tid >> 5;
  const int col  = lane & 15;
  const int mrow = (lane >> 4) * 8;
  const int row0 = blockIdx.x * ROWS;

  if (tid < ROWS) sx[tid] = (row0 + tid < nRows) ? ind[row0 + tid] : 0.f;
  __syncthreads();
  for (int i = tid; i < ROWS * H1; i += 256) {
    int r = i / H1, c = i % H1;
    sH[r][c] = (__bf16)fmaxf(sx[r] * w1[c] + b1[c], 0.f);
  }
  __syncthreads();

  {
    const int n0 = wave * 16;
    v8f acc[4] = {};
#pragma unroll
    for (int ks = 0; ks < H1 / 32; ++ks) {
      v16bf b = load_B_t(W2t + (size_t)n0 * H1 + ks * 32, H1, lane);
#pragma unroll
      for (int rt = 0; rt < 4; ++rt) {
        v16bf a = load_A_lds(&sH[rt * 16][0] + ks * 32, H1, lane);
        acc[rt] = wmma_bf16(a, b, acc[rt]);
      }
    }
    const float bias = b2[n0 + col];
#pragma unroll
    for (int rt = 0; rt < 4; ++rt)
#pragma unroll
      for (int r = 0; r < 8; ++r) {
        int g = row0 + rt * 16 + mrow + r;
        if (g < nRows)
          out[(size_t)g * D + n0 + col] = (__bf16)fmaxf(acc[rt][r] + bias, 0.f);
      }
  }
}

// ---------------------------------------------------------------------------
// One edge MLP pass over a 64-edge tile.  X = [blk0|blk1|blk2] (64x384).
// Each B fragment is reused across 4 row tiles -> 4x less L2 weight traffic.
// ---------------------------------------------------------------------------
__device__ __forceinline__ void edge_pass(
    const __bf16 (*blk0)[D], const __bf16 (*blk1)[D], const __bf16 (*blk2)[D],
    __bf16 (*sH)[H1],
    const __bf16* __restrict__ W1t, const float* __restrict__ b1v,
    const __bf16* __restrict__ W2t, const float* __restrict__ b2v,
    const int* sNode, float* __restrict__ accum,
    int e0, int nEdges, int lane, int wave) {
  const int col  = lane & 15;
  const int mrow = (lane >> 4) * 8;

  // layer 1: [64,256]; wave: 2 col tiles x 4 row tiles; K=384 (12 steps)
#pragma unroll
  for (int tc = 0; tc < 2; ++tc) {
    const int n0 = (wave * 2 + tc) * 16;
    v8f acc[4] = {};
#pragma unroll
    for (int ks = 0; ks < 12; ++ks) {
      const __bf16 (*blk)[D] = (ks < 4) ? blk0 : (ks < 8) ? blk1 : blk2;
      v16bf b = load_B_t(W1t + (size_t)n0 * D3 + ks * 32, D3, lane);
#pragma unroll
      for (int rt = 0; rt < 4; ++rt) {
        v16bf a = load_A_lds(&blk[rt * 16][0] + (ks & 3) * 32, D, lane);
        acc[rt] = wmma_bf16(a, b, acc[rt]);
      }
    }
    const float bias = b1v[n0 + col];
#pragma unroll
    for (int rt = 0; rt < 4; ++rt)
#pragma unroll
      for (int r = 0; r < 8; ++r)
        sH[rt * 16 + mrow + r][n0 + col] = (__bf16)fmaxf(acc[rt][r] + bias, 0.f);
  }
  __syncthreads();

  // layer 2: [64,128]; wave: 1 col tile x 4 row tiles; K=256; atomic scatter
  {
    const int n0 = wave * 16;
    v8f acc[4] = {};
#pragma unroll
    for (int ks = 0; ks < H1 / 32; ++ks) {
      v16bf b = load_B_t(W2t + (size_t)n0 * H1 + ks * 32, H1, lane);
#pragma unroll
      for (int rt = 0; rt < 4; ++rt) {
        v16bf a = load_A_lds(&sH[rt * 16][0] + ks * 32, H1, lane);
        acc[rt] = wmma_bf16(a, b, acc[rt]);
      }
    }
    const float bias = b2v[n0 + col];
#pragma unroll
    for (int rt = 0; rt < 4; ++rt)
#pragma unroll
      for (int r = 0; r < 8; ++r) {
        const int lrow = rt * 16 + mrow + r;
        const int e = e0 + lrow;
        if (e < nEdges) {
          const float v = fmaxf(acc[rt][r] + bias, 0.f);
          atomicAdd(&accum[(size_t)sNode[lrow] * D + n0 + col], v);
        }
      }
  }
}

// ---------------------------------------------------------------------------
// Edge kernel: 64-edge tile.  Dynamic LDS (hardware-legal up to 320KB/WG):
//   [ sHp 64x128 | sHc 64x128 | sEx 64x128 | sH 64x256 ]  (bf16) = 80 KB
// sEx holds edge_p features for the P pass, then is reloaded with edge_c.
// ---------------------------------------------------------------------------
__global__ __launch_bounds__(256)
void k_edge(const __bf16* __restrict__ hbf,
            const int* __restrict__ pidx, const int* __restrict__ cidx,
            const __bf16* __restrict__ ep, const __bf16* __restrict__ ec,
            const __bf16* __restrict__ Pw1t, const float* __restrict__ Pb1,
            const __bf16* __restrict__ Pw2t, const float* __restrict__ Pb2,
            const __bf16* __restrict__ Cw1t, const float* __restrict__ Cb1,
            const __bf16* __restrict__ Cw2t, const float* __restrict__ Cb2,
            float* __restrict__ spa, float* __restrict__ sca, int nEdges) {
  extern __shared__ __align__(16) char smem[];
  __bf16 (*sHp)[D]  = (__bf16 (*)[D])smem;
  __bf16 (*sHc)[D]  = sHp + ROWS;
  __bf16 (*sEx)[D]  = sHc + ROWS;
  __bf16 (*sH)[H1]  = (__bf16 (*)[H1])(smem + (size_t)3 * ROWS * D * 2);
  __shared__ int sPn[ROWS], sCn[ROWS];

  const int tid  = threadIdx.x;
  const int lane = tid & 31;
  const int wave = tid >> 5;
  const int e0   = blockIdx.x * ROWS;

  if (tid < ROWS) {
    int e = min(e0 + tid, nEdges - 1);
    sPn[tid] = pidx[e];
    sCn[tid] = cidx[e];
  }
  __syncthreads();

  // warm L2 for the random h-row gathers (global_prefetch_b8)
  if (tid < ROWS) {
    __builtin_prefetch(hbf + (size_t)sPn[tid] * D, 0, 0);
    __builtin_prefetch(hbf + (size_t)sCn[tid] * D, 0, 0);
  }

  // vectorized gather: 16B chunks, b128 loads / ds b128 stores
  for (int i = tid; i < ROWS * 16; i += 256) {
    int r = i >> 4, c4 = i & 15;
    int e = e0 + r;
    uint4 z = {0u, 0u, 0u, 0u};
    uint4 vp = z, vc = z, vx = z;
    if (e < nEdges) {
      vp = ((const uint4*)(hbf + (size_t)sPn[r] * D))[c4];
      vc = ((const uint4*)(hbf + (size_t)sCn[r] * D))[c4];
      vx = ((const uint4*)(ep + (size_t)e * D))[c4];
    }
    ((uint4*)&sHp[r][0])[c4] = vp;
    ((uint4*)&sHc[r][0])[c4] = vc;
    ((uint4*)&sEx[r][0])[c4] = vx;
  }
  __syncthreads();

  // s_p = MLP_P([hc, hp, edge_p]) -> scatter by edge_p_node
  edge_pass(sHc, sHp, sEx, sH, Pw1t, Pb1, Pw2t, Pb2, sPn, spa, e0, nEdges, lane, wave);
  __syncthreads();

  // reload sEx with edge_c features
  for (int i = tid; i < ROWS * 16; i += 256) {
    int r = i >> 4, c4 = i & 15;
    int e = e0 + r;
    uint4 vx = {0u, 0u, 0u, 0u};
    if (e < nEdges) vx = ((const uint4*)(ec + (size_t)e * D))[c4];
    ((uint4*)&sEx[r][0])[c4] = vx;
  }
  __syncthreads();

  // s_c = MLP_C([hp, hc, edge_c]) -> scatter by edge_c_node
  edge_pass(sHp, sHc, sEx, sH, Cw1t, Cb1, Cw2t, Cb2, sCn, sca, e0, nEdges, lane, wave);
}

// ---------------------------------------------------------------------------
// Node kernel: normalize s_p/s_c + mask*token, MLP_A on [h|s_p|s_c], residual.
// Dynamic LDS: [ sX 64x384 bf16 | sH 64x256 bf16 | sHf 64x128 f32 ] = 112 KB
// ---------------------------------------------------------------------------
__global__ __launch_bounds__(256)
void k_node(const float* __restrict__ h,
            const float* __restrict__ spa, const float* __restrict__ sca,
            const float* __restrict__ invp, const float* __restrict__ invc,
            const float* __restrict__ pmask, const float* __restrict__ cmask,
            const float* __restrict__ stok, const float* __restrict__ etok,
            const __bf16* __restrict__ Aw1t, const float* __restrict__ Ab1,
            const __bf16* __restrict__ Aw2t, const float* __restrict__ Ab2,
            float* __restrict__ h_out, __bf16* __restrict__ hbf_out, int nRows) {
  extern __shared__ __align__(16) char smem[];
  __bf16 (*sX)[D3] = (__bf16 (*)[D3])smem;
  __bf16 (*sH)[H1] = (__bf16 (*)[H1])(smem + (size_t)ROWS * D3 * 2);
  float  (*sHf)[D] = (float  (*)[D])(smem + (size_t)ROWS * D3 * 2 + (size_t)ROWS * H1 * 2);

  const int tid  = threadIdx.x;
  const int lane = tid & 31;
  const int wave = tid >> 5;
  const int col  = lane & 15;
  const int mrow = (lane >> 4) * 8;
  const int row0 = blockIdx.x * ROWS;

  for (int i = tid; i < ROWS * (D / 4); i += 256) {
    int r = i / (D / 4), c4 = i % (D / 4);
    int g = row0 + r;
    int c = c4 * 4;
    float4 hv = {0.f, 0.f, 0.f, 0.f}, sp = hv, sc = hv;
    if (g < nRows) {
      hv = ((const float4*)(h + (size_t)g * D))[c4];
      float4 spv = ((const float4*)(spa + (size_t)g * D))[c4];
      float4 scv = ((const float4*)(sca + (size_t)g * D))[c4];
      float ip = invp[g], ic = invc[g], pm = pmask[g], cm = cmask[g];
      sp.x = spv.x * ip + pm * stok[c + 0];  sc.x = scv.x * ic + cm * etok[c + 0];
      sp.y = spv.y * ip + pm * stok[c + 1];  sc.y = scv.y * ic + cm * etok[c + 1];
      sp.z = spv.z * ip + pm * stok[c + 2];  sc.z = scv.z * ic + cm * etok[c + 2];
      sp.w = spv.w * ip + pm * stok[c + 3];  sc.w = scv.w * ic + cm * etok[c + 3];
    }
    ((float4*)&sHf[r][0])[c4] = hv;
    sX[r][c + 0] = (__bf16)hv.x;  sX[r][c + 1] = (__bf16)hv.y;
    sX[r][c + 2] = (__bf16)hv.z;  sX[r][c + 3] = (__bf16)hv.w;
    sX[r][D + c + 0] = (__bf16)sp.x;  sX[r][D + c + 1] = (__bf16)sp.y;
    sX[r][D + c + 2] = (__bf16)sp.z;  sX[r][D + c + 3] = (__bf16)sp.w;
    sX[r][2 * D + c + 0] = (__bf16)sc.x;  sX[r][2 * D + c + 1] = (__bf16)sc.y;
    sX[r][2 * D + c + 2] = (__bf16)sc.z;  sX[r][2 * D + c + 3] = (__bf16)sc.w;
  }
  __syncthreads();

  // layer 1: [64,256]; K=384 contiguous in sX
#pragma unroll
  for (int tc = 0; tc < 2; ++tc) {
    const int n0 = (wave * 2 + tc) * 16;
    v8f acc[4] = {};
#pragma unroll
    for (int ks = 0; ks < 12; ++ks) {
      v16bf b = load_B_t(Aw1t + (size_t)n0 * D3 + ks * 32, D3, lane);
#pragma unroll
      for (int rt = 0; rt < 4; ++rt) {
        v16bf a = load_A_lds(&sX[rt * 16][0] + ks * 32, D3, lane);
        acc[rt] = wmma_bf16(a, b, acc[rt]);
      }
    }
    const float bias = Ab1[n0 + col];
#pragma unroll
    for (int rt = 0; rt < 4; ++rt)
#pragma unroll
      for (int r = 0; r < 8; ++r)
        sH[rt * 16 + mrow + r][n0 + col] = (__bf16)fmaxf(acc[rt][r] + bias, 0.f);
  }
  __syncthreads();

  // layer 2: [64,128]; K=256; h_new = relu(h + relu(out))
  {
    const int n0 = wave * 16;
    v8f acc[4] = {};
#pragma unroll
    for (int ks = 0; ks < H1 / 32; ++ks) {
      v16bf b = load_B_t(Aw2t + (size_t)n0 * H1 + ks * 32, H1, lane);
#pragma unroll
      for (int rt = 0; rt < 4; ++rt) {
        v16bf a = load_A_lds(&sH[rt * 16][0] + ks * 32, H1, lane);
        acc[rt] = wmma_bf16(a, b, acc[rt]);
      }
    }
    const float bias = Ab2[n0 + col];
#pragma unroll
    for (int rt = 0; rt < 4; ++rt)
#pragma unroll
      for (int r = 0; r < 8; ++r) {
        int g = row0 + rt * 16 + mrow + r;
        if (g < nRows) {
          float v  = fmaxf(acc[rt][r] + bias, 0.f);
          float hn = fmaxf(sHf[rt * 16 + mrow + r][n0 + col] + v, 0.f);
          h_out  [(size_t)g * D + n0 + col] = hn;
          hbf_out[(size_t)g * D + n0 + col] = (__bf16)hn;
        }
      }
  }
}

// ---------------------------------------------------------------------------
extern "C" void kernel_launch(void* const* d_in, const int* in_sizes, int n_in,
                              void* d_out, int out_size, void* d_ws, size_t ws_size,
                              hipStream_t stream) {
  (void)n_in; (void)out_size; (void)ws_size;
  const float* batch_token = (const float*)d_in[0];
  const int*   edge_p_node = (const int*)d_in[1];
  const int*   edge_c_node = (const int*)d_in[2];
  const float* edge_p_ind  = (const float*)d_in[3];
  const float* edge_c_ind  = (const float*)d_in[4];
  const float* p_mask      = (const float*)d_in[5];
  const float* c_mask      = (const float*)d_in[6];
  const float* start_token = (const float*)d_in[7];
  const float* end_token   = (const float*)d_in[8];
  const float* Vw1 = (const float*)d_in[9];  const float* Vb1 = (const float*)d_in[10];
  const float* Vw2 = (const float*)d_in[11]; const float* Vb2 = (const float*)d_in[12];
  const float* Ew1 = (const float*)d_in[13]; const float* Eb1 = (const float*)d_in[14];
  const float* Ew2 = (const float*)d_in[15]; const float* Eb2 = (const float*)d_in[16];
  const float* Pw1 = (const float*)d_in[17]; const float* Pb1 = (const float*)d_in[18];
  const float* Pw2 = (const float*)d_in[19]; const float* Pb2 = (const float*)d_in[20];
  const float* Cw1 = (const float*)d_in[21]; const float* Cb1 = (const float*)d_in[22];
  const float* Cw2 = (const float*)d_in[23]; const float* Cb2 = (const float*)d_in[24];
  const float* Aw1 = (const float*)d_in[25]; const float* Ab1 = (const float*)d_in[26];
  const float* Aw2 = (const float*)d_in[27]; const float* Ab2 = (const float*)d_in[28];

  const int N = in_sizes[0] / D;   // 50000
  const int E = in_sizes[1];       // 500000

  char* ws = (char*)d_ws;
  size_t off = 0;
  auto alloc = [&](size_t bytes) -> char* {
    off = (off + 255) & ~(size_t)255;
    char* p = ws + off;
    off += bytes;
    return p;
  };
  __bf16* Vw1t = (__bf16*)alloc((size_t)D  * H1 * 2);
  __bf16* Vw2t = (__bf16*)alloc((size_t)H1 * D  * 2);
  __bf16* Ew2t = (__bf16*)alloc((size_t)H1 * D  * 2);
  __bf16* Pw1t = (__bf16*)alloc((size_t)D3 * H1 * 2);
  __bf16* Pw2t = (__bf16*)alloc((size_t)H1 * D  * 2);
  __bf16* Cw1t = (__bf16*)alloc((size_t)D3 * H1 * 2);
  __bf16* Cw2t = (__bf16*)alloc((size_t)H1 * D  * 2);
  __bf16* Aw1t = (__bf16*)alloc((size_t)D3 * H1 * 2);
  __bf16* Aw2t = (__bf16*)alloc((size_t)H1 * D  * 2);
  float*  h_f32 = (float*)alloc((size_t)N * D * 4);
  __bf16* h_bf  = (__bf16*)alloc((size_t)N * D * 2);
  __bf16* ep    = (__bf16*)alloc((size_t)E * D * 2);
  __bf16* ec    = (__bf16*)alloc((size_t)E * D * 2);
  float*  spa   = (float*)alloc((size_t)N * D * 4 * 2);  // spa | sca
  float*  sca   = spa + (size_t)N * D;
  float*  cnt   = (float*)alloc((size_t)N * 4 * 2);      // cp | cc
  float*  cp = cnt;
  float*  cc = cnt + N;

  auto cvt = [&](const float* W, __bf16* Wt, int K, int Ncols) {
    int total = K * Ncols;
    k_convert_w<<<(total + 255) / 256, 256, 0, stream>>>(W, Wt, K, Ncols);
  };
  cvt(Vw1, Vw1t, D,  H1);  cvt(Vw2, Vw2t, H1, D);
  cvt(Ew2, Ew2t, H1, D);
  cvt(Pw1, Pw1t, D3, H1);  cvt(Pw2, Pw2t, H1, D);
  cvt(Cw1, Cw1t, D3, H1);  cvt(Cw2, Cw2t, H1, D);
  cvt(Aw1, Aw1t, D3, H1);  cvt(Aw2, Aw2t, H1, D);

  k_zero <<<((size_t)2 * N + 255) / 256, 256, 0, stream>>>(cnt, (long)2 * N);
  k_count<<<(E + 255) / 256, 256, 0, stream>>>(edge_p_node, edge_c_node, cp, cc, E);
  k_inv  <<<(N + 255) / 256, 256, 0, stream>>>(cp, cc, N);

  const int nWGn = (N + ROWS - 1) / ROWS;
  const int nWGe = (E + ROWS - 1) / ROWS;

  k_mlp_v<<<nWGn, 256, 0, stream>>>(batch_token, Vw1t, Vb1, Vw2t, Vb2, h_f32, h_bf, N);
  k_mlp_e<<<nWGe, 256, 0, stream>>>(edge_p_ind, Ew1, Eb1, Ew2t, Eb2, ep, E);
  k_mlp_e<<<nWGe, 256, 0, stream>>>(edge_c_ind, Ew1, Eb1, Ew2t, Eb2, ec, E);

  const size_t edge_smem = ((size_t)3 * ROWS * D + (size_t)ROWS * H1) * 2;      // 80 KB
  const size_t node_smem = ((size_t)ROWS * D3 + (size_t)ROWS * H1) * 2
                         + (size_t)ROWS * D * 4;                                 // 112 KB

  for (int hop = 0; hop < NUM_HOPS - 1; ++hop) {
    k_zero<<<(((size_t)2 * N * D) + 255) / 256, 256, 0, stream>>>(spa, (long)2 * N * D);
    k_edge<<<nWGe, 256, edge_smem, stream>>>(h_bf, edge_p_node, edge_c_node, ep, ec,
                                             Pw1t, Pb1, Pw2t, Pb2,
                                             Cw1t, Cb1, Cw2t, Cb2, spa, sca, E);
    float* hout = (hop == NUM_HOPS - 2) ? (float*)d_out : h_f32;
    k_node<<<nWGn, 256, node_smem, stream>>>(h_f32, spa, sca, cp, cc,
                                             p_mask, c_mask, start_token, end_token,
                                             Aw1t, Ab1, Aw2t, Ab2, hout, h_bf, N);
  }
}
